// SpikingReservoir_55465207660615
// MI455X (gfx1250) — matrix operand order repaired and checked by reading, not compiled, not run on hardware
//
#include <hip/hip_runtime.h>

// ---- CDNA5 WMMA types ----
typedef __attribute__((ext_vector_type(16))) __bf16 v16bf;
typedef __attribute__((ext_vector_type(8)))  float  v8f;

#define NRES    512
#define NIN     8
#define TSTEPS  3000
#define SROW    552              // spike row: 512 spikes + 8 x_t + 24 zeros + pad
                                 // (1104B rows: 16B aligned, row starts at 20*r%64 banks)
#define WBROW   520              // weight row: 512 + 8 pad (1040B, 4*k%64 bank starts)
#define KLDS    9                // 9 weight K-chunks (k=0..287) staged in LDS
#define KREGW   7                // 7 weight K-chunks (k=288..511) in VGPRs
// + 1 input-projection chunk (W_in^T, K=512..543) in VGPRs -> 17 WMMAs/step

__device__ __forceinline__ unsigned short f2bf_rne(float f) {
    union { float f; unsigned u; } x; x.f = f;
    unsigned r = x.u + 0x7FFFu + ((x.u >> 16) & 1u);
    return (unsigned short)(r >> 16);
}

// WT[k][r] = bf16(W_rec[r][k])  (512x512, once; L2-resident afterwards)
__global__ void wrec_transpose_bf16(const float* __restrict__ W,
                                    unsigned short* __restrict__ WT) {
    int idx = blockIdx.x * blockDim.x + threadIdx.x;   // 0 .. 512*512-1
    int r = idx & (NRES - 1);
    int k = idx >> 9;
    WT[idx] = f2bf_rne(W[r * NRES + k]);
}

// WIT[k][r] = bf16(W_in[r][k]) for k<8, else 0   (32x512 zero-padded input block)
__global__ void win_transpose_bf16(const float* __restrict__ Win,
                                   unsigned short* __restrict__ WIT) {
    int idx = blockIdx.x * blockDim.x + threadIdx.x;   // 0 .. 32*512-1
    int r = idx & (NRES - 1);
    int k = idx >> 9;
    WIT[idx] = (k < NIN) ? f2bf_rne(Win[r * NIN + k]) : (unsigned short)0;
}

// Persistent per-batch-chunk reservoir scan.
// grid = 4 blocks (16 batch rows each), block = 1024 threads = 32 waves.
// Wave w owns output tile: rows b = wb*16..+15, cols r = w*16..+15.
// i_in = [h | x_t] @ [W_rec^T ; W_in^T] : K = 544, 17 x v_wmma_f32_16x16x32_bf16.
__launch_bounds__(1024, 1)
__global__ void reservoir_scan(const float* __restrict__ x,            // [64][8][3000]
                               const unsigned short* __restrict__ WT,  // [512][512] bf16
                               const unsigned short* __restrict__ WIT, // [32][512]  bf16
                               float* __restrict__ out) {              // [64][512]
    __shared__ unsigned short S[16 * SROW];       // [h | x_t] matrix (bf16)   ~17.7 KB
    __shared__ unsigned short WB[KLDS * 32 * WBROW]; // W_rec^T k=0..287       ~299.5 KB

    const int tid  = threadIdx.x;
    const int lane = tid & 31;
    const int wave = tid >> 5;                // 0..31
    const int wb   = blockIdx.x;              // batch chunk
    const int half = lane >> 4;               // 0 or 1
    const int n0   = wave * 16;
    const int rcol = n0 + (lane & 15);        // this lane's reservoir column (constant)

    // zero-init A matrix (h0 = 0, zero padding columns)
    for (int i = tid; i < 16 * SROW; i += 1024) S[i] = 0;

    // stage W_rec^T rows k=0..287 into LDS (padded rows), 16B copies
    for (int idx = tid; idx < KLDS * 32 * (NRES / 8); idx += 1024) {
        int row = idx >> 6;            // k
        int q   = idx & 63;            // 8-element group within row
        *(uint4*)(WB + row * WBROW + q * 8) = *(const uint4*)(WT + row * NRES + q * 8);
    }

    // register-resident B: 7 weight chunks (k=288..511) + 1 input chunk (W_in^T)
    v16bf Breg[KREGW + 1];
#pragma unroll
    for (int i = 0; i < KREGW; ++i) {
        const unsigned short* bp = WT + (size_t)(KLDS * 32 + i * 32 + half * 8) * NRES + n0 + (lane & 15);
        union { unsigned short e[16]; v16bf v; } tmp;
        for (int s = 0; s < 8; ++s) {
            tmp.e[s]     = bp[(size_t)s * NRES];
            tmp.e[s + 8] = bp[(size_t)(16 + s) * NRES];
        }
        Breg[i] = tmp.v;
    }
    {
        const unsigned short* bp = WIT + (size_t)(half * 8) * NRES + n0 + (lane & 15);
        union { unsigned short e[16]; v16bf v; } tmp;
        for (int s = 0; s < 8; ++s) {
            tmp.e[s]     = bp[(size_t)s * NRES];
            tmp.e[s + 8] = bp[(size_t)(16 + s) * NRES];
        }
        Breg[KREGW] = tmp.v;
    }

    // preload x columns for t = 0 into A matrix (cols 512..519)
    if (tid < 16 * NIN) {
        int bl = tid >> 3, i = tid & 7;
        S[bl * SROW + NRES + i] = f2bf_rne(x[((wb * 16 + bl) * NIN + i) * TSTEPS + 0]);
    }

    v8f v   = {};   // membrane potential, 8 batch rows per lane (C-matrix layout)
    v8f acc = {};   // spike counts

    // A: lane reads row (lane&15), chunks at k0 + half*8 and k0 + 16 + half*8
    const unsigned short* Sbase = &S[(lane & 15) * SROW + half * 8];
    // B (LDS part): lane reads WB col n0 + (lane&15), rows k0 + half*8 and k0 + 16 + half*8
    const unsigned short* Wbase = &WB[half * 8 * WBROW + n0 + (lane & 15)];

    __syncthreads();   // S zeroed + x(0) staged, WB staged

    for (int t = 0; t < TSTEPS; ++t) {
        // i_in tile = [h|x_t] @ [W_rec^T;W_in^T] : 17 WMMAs over K=544
        v8f c = {};
#pragma unroll
        for (int cc = 0; cc < KLDS; ++cc) {       // K in LDS
            const int k0 = cc * 32;
            union { uint4 u[2]; v16bf v; } A;
            A.u[0] = *(const uint4*)(Sbase + k0);
            A.u[1] = *(const uint4*)(Sbase + k0 + 16);
            union { unsigned short e[16]; v16bf v; } B;
            const unsigned short* bp = Wbase + (size_t)k0 * WBROW;
#pragma unroll
            for (int s = 0; s < 8; ++s) {
                B.e[s]     = bp[s * WBROW];
                B.e[s + 8] = bp[(16 + s) * WBROW];
            }
            c = __builtin_amdgcn_wmma_f32_16x16x32_bf16(
                    false, A.v, false, B.v, (short)0, c, false, false);
        }
#pragma unroll
        for (int i = 0; i < KREGW + 1; ++i) {     // K in registers (incl. input chunk)
            const int k0 = (KLDS + i) * 32;
            union { uint4 u[2]; v16bf v; } A;
            A.u[0] = *(const uint4*)(Sbase + k0);
            A.u[1] = *(const uint4*)(Sbase + k0 + 16);
            c = __builtin_amdgcn_wmma_f32_16x16x32_bf16(
                    false, A.v, false, Breg[i], (short)0, c, false, false);
        }

        // LIF update for this lane's 8 (b, r) elements
        float snew[8];
#pragma unroll
        for (int j = 0; j < 8; ++j) {
            float vv = 0.5f * (v[j] + c[j]);                   // v += (i_in - v)/TAU
            float s  = (vv - 1.0f >= 0.0f) ? 1.0f : 0.0f;      // Heaviside spike
            v[j]   = vv * (1.0f - s);                          // hard reset to 0
            acc[j] += s;
            snew[j] = s;
        }

        __syncthreads();   // all waves done reading S

        // write new spikes; stage next x slice into A-matrix columns 512..519
#pragma unroll
        for (int j = 0; j < 8; ++j) {
            const int bl = j + 8 * half;
            S[bl * SROW + rcol] = (snew[j] != 0.0f) ? (unsigned short)0x3F80
                                                    : (unsigned short)0;
        }
        if (t + 1 < TSTEPS && tid < 16 * NIN) {
            int bl = tid >> 3, i = tid & 7;
            S[bl * SROW + NRES + i] =
                f2bf_rne(x[((wb * 16 + bl) * NIN + i) * TSTEPS + (t + 1)]);
        }
        __syncthreads();   // spikes + next x visible before next GEMM
    }

    // spike_counts / T
    const float inv_t = 1.0f / (float)TSTEPS;
#pragma unroll
    for (int j = 0; j < 8; ++j) {
        const int b = wb * 16 + j + 8 * half;
        out[b * NRES + rcol] = acc[j] * inv_t;
    }
}

extern "C" void kernel_launch(void* const* d_in, const int* in_sizes, int n_in,
                              void* d_out, int out_size, void* d_ws, size_t ws_size,
                              hipStream_t stream) {
    const float* x     = (const float*)d_in[0];   // [64][8][3000]
    const float* W_in  = (const float*)d_in[1];   // [512][8]
    const float* W_rec = (const float*)d_in[2];   // [512][512]
    float* out = (float*)d_out;                   // [64][512]

    unsigned short* WT  = (unsigned short*)d_ws;           // 512*512 bf16 = 512 KB
    unsigned short* WIT = WT + (size_t)NRES * NRES;        // 32*512  bf16 =  32 KB

    // 1) transpose + bf16-convert weights (once; stay hot in L2)
    wrec_transpose_bf16<<<(NRES * NRES) / 256, 256, 0, stream>>>(W_rec, WT);
    win_transpose_bf16<<<(32 * NRES) / 256, 256, 0, stream>>>(W_in, WIT);

    // 2) persistent scan: 4 independent batch chunks, one workgroup (32 waves) each
    reservoir_scan<<<4, 1024, 0, stream>>>(x, WT, WIT, out);
}
